// BasePolicy_51616916964031
// MI455X (gfx1250) — compile-verified, matrix operand order checked by reading, not run on hardware
//
#include <hip/hip_runtime.h>
#include <stdint.h>

typedef __attribute__((ext_vector_type(16))) _Float16 v16h;
typedef __attribute__((ext_vector_type(8)))  float    v8f;

#define DEV static __device__ __forceinline__

static constexpr int H = 128;      // hidden
static constexpr int DI = 20;      // intruder features
static constexpr int DG = 512;     // grid features
static constexpr int DOWN = 16;    // own features

// workspace layout (32-bit word offsets)
static constexpr int OW_WOWN  = 0;              // [128][16w]  (K=32 padded)  f16
static constexpr int OW_WINTR = 2048;           // [128][16w]
static constexpr int OW_WQ    = 4096;           // [128][64w]  (K=128)
static constexpr int OW_WK    = 12288;          // [128][64w]
static constexpr int OW_WV    = 20480;          // [128][64w]
static constexpr int OW_WG    = 28672;          // [128][256w] (K=512)
static constexpr int OW_WC    = 61440;          // f32 [384][4]  = W1@W2
static constexpr int OW_BC    = 62976;          // f32 [4]       = b1@W2+b2
static constexpr int OW_DYN   = 63488;          // f32 own_e[B][128], q[B][128], grid[B][128]

DEV uint32_t packh(float a, float b) {
  union { _Float16 h[2]; uint32_t u; } x;
  x.h[0] = (_Float16)a; x.h[1] = (_Float16)b;
  return x.u;
}
DEV uint16_t f2h16(float a) {
  union { _Float16 h; uint16_t u; } x;
  x.h = (_Float16)a; return x.u;
}

DEV v8f vzero() {
  v8f z;
#pragma unroll
  for (int i = 0; i < 8; ++i) z[i] = 0.f;
  return z;
}

// Load a 16x32 f16 fragment (A layout; B uses same per-lane map on the
// pre-transposed [h][K] weight buffers). buf = row-major [rows][rs words],
// kw = word offset of the K-chunk (16 words = 32 f16). ISA 7.12.2:
// lane m=l&15 half=l>>4; VGPR j<4 -> K=8*half+2j ; j>=4 -> K=16+8*half+2(j-4)
DEV v16h frag_ld(const uint32_t* buf, int rs, int row0, int kw, int lane) {
  int m = lane & 15, half = lane >> 4;
  const uint32_t* p = buf + (size_t)(row0 + m) * rs + kw + 4 * half;
  union { v16h v; uint4 q[2]; } r;
  r.q[0] = *reinterpret_cast<const uint4*>(p);
  r.q[1] = *reinterpret_cast<const uint4*>(p + 8);
  return r.v;
}

DEV v8f wmma16(v16h a, v16h b, v8f c) {
  return __builtin_amdgcn_wmma_f32_16x16x32_f16(false, a, false, b, (short)0, c, false, false);
}

// ---------------------------------------------------------------- prep
__global__ __launch_bounds__(256) void prep_kernel(
    const float* __restrict__ W_own, const float* __restrict__ W_intr,
    const float* __restrict__ Wq, const float* __restrict__ Wk,
    const float* __restrict__ Wv, const float* __restrict__ W_grid,
    const float* __restrict__ W1, const float* __restrict__ b1,
    const float* __restrict__ W2, const float* __restrict__ b2,
    uint32_t* __restrict__ wsw) {
  const int tid = threadIdx.x; const int NT = 256;
  for (int i = tid; i < 128 * 16; i += NT) {            // W_ownT [128][32 pad]
    int h = i >> 4, p = i & 15, k0 = 2 * p;
    float f0 = (k0 < DOWN) ? W_own[k0 * H + h] : 0.f;
    float f1 = (k0 + 1 < DOWN) ? W_own[(k0 + 1) * H + h] : 0.f;
    wsw[OW_WOWN + h * 16 + p] = packh(f0, f1);
  }
  for (int i = tid; i < 128 * 16; i += NT) {            // W_intrT [128][32 pad]
    int h = i >> 4, p = i & 15, k0 = 2 * p;
    float f0 = (k0 < DI) ? W_intr[k0 * H + h] : 0.f;
    float f1 = (k0 + 1 < DI) ? W_intr[(k0 + 1) * H + h] : 0.f;
    wsw[OW_WINTR + h * 16 + p] = packh(f0, f1);
  }
  for (int i = tid; i < 128 * 64; i += NT) {            // WqT/WkT/WvT [128][128]
    int h = i >> 6, p = i & 63, k0 = 2 * p;
    wsw[OW_WQ + h * 64 + p] = packh(Wq[k0 * H + h], Wq[(k0 + 1) * H + h]);
    wsw[OW_WK + h * 64 + p] = packh(Wk[k0 * H + h], Wk[(k0 + 1) * H + h]);
    wsw[OW_WV + h * 64 + p] = packh(Wv[k0 * H + h], Wv[(k0 + 1) * H + h]);
  }
  for (int i = tid; i < 128 * 256; i += NT) {           // W_gridT [128][512]
    int h = i >> 8, p = i & 255, k0 = 2 * p;
    wsw[OW_WG + h * 256 + p] = packh(W_grid[k0 * H + h], W_grid[(k0 + 1) * H + h]);
  }
  float* wsf = (float*)wsw;                             // Wc = W1@W2, bc = b1@W2+b2
  for (int r = tid; r < 384; r += NT) {
    float acc[4] = {0.f, 0.f, 0.f, 0.f};
    for (int k = 0; k < 128; ++k) {
      float w = W1[r * 128 + k];
#pragma unroll
      for (int j = 0; j < 4; ++j) acc[j] += w * W2[k * 4 + j];
    }
#pragma unroll
    for (int j = 0; j < 4; ++j) wsf[OW_WC + r * 4 + j] = acc[j];
  }
  if (tid < 4) {
    float s = b2[tid];
    for (int k = 0; k < 128; ++k) s += b1[k] * W2[k * 4 + tid];
    wsf[OW_BC + tid] = s;
  }
}

// ---------------------------------------------------------------- embed
// one block per 16 batch rows: own_e = relu(s0@W_own+b), q = own_e@Wq,
// own_grid = relu(s1@W_grid+b)
__global__ __launch_bounds__(128) void embed_kernel(
    const float* __restrict__ s0, const float* __restrict__ s1,
    const float* __restrict__ b_own, const float* __restrict__ b_grid,
    const uint32_t* __restrict__ wown, const uint32_t* __restrict__ wq,
    const uint32_t* __restrict__ wg, float* __restrict__ own_e,
    float* __restrict__ q_out, float* __restrict__ grid_out) {
  __shared__ alignas(16) uint32_t a0h[16 * 20];     // s0 tile f16 K=32 pad
  __shared__ alignas(16) uint32_t wol[128 * 20];
  __shared__ alignas(16) uint32_t oeh[16 * 68];     // own_e f16 K=128
  __shared__ alignas(16) uint32_t wql[128 * 68];
  __shared__ alignas(16) uint32_t s1h[16 * 260];    // s1 tile f16 K=512
  __shared__ alignas(16) uint32_t wgl[128 * 68];    // W_gridT K-chunk buffer

  const int tid = threadIdx.x, lane = tid & 31, wid = tid >> 5;
  const int r0 = blockIdx.x * 16;
  const int half = lane >> 4, col = lane & 15;

  for (int i = tid; i < 16 * 16; i += 128) {
    int r = i >> 4, p = i & 15, k0 = 2 * p;
    float f0 = (k0 < DOWN) ? s0[(size_t)(r0 + r) * DOWN + k0] : 0.f;
    float f1 = (k0 + 1 < DOWN) ? s0[(size_t)(r0 + r) * DOWN + k0 + 1] : 0.f;
    a0h[r * 20 + p] = packh(f0, f1);
  }
  for (int i = tid; i < 128 * 16; i += 128) { int h = i >> 4, p = i & 15; wol[h * 20 + p] = wown[i]; }
  for (int i = tid; i < 128 * 64; i += 128) { int h = i >> 6, p = i & 63; wql[h * 68 + p] = wq[i]; }
  for (int i = tid; i < 16 * 256; i += 128) {
    int r = i >> 8, p = i & 255, k0 = 2 * p;
    s1h[r * 260 + p] = packh(s1[(size_t)(r0 + r) * DG + k0], s1[(size_t)(r0 + r) * DG + k0 + 1]);
  }
  __syncthreads();

  // own_e: M=16, K=32, 8 H-tiles over 4 waves
  uint16_t* oe16 = reinterpret_cast<uint16_t*>(oeh);
  {
    v16h a = frag_ld(a0h, 20, 0, 0, lane);
#pragma unroll
    for (int t = 0; t < 2; ++t) {
      int ht = wid + 4 * t;
      v8f acc = vzero();
      acc = wmma16(a, frag_ld(wol, 20, ht * 16, 0, lane), acc);
      int h = ht * 16 + col;
      float bias = b_own[h];
#pragma unroll
      for (int r = 0; r < 8; ++r) {
        int row = r + 8 * half;
        float v = acc[r] + bias; v = v > 0.f ? v : 0.f;
        own_e[(size_t)(r0 + row) * H + h] = v;
        oe16[row * 136 + h] = f2h16(v);
      }
    }
  }
  __syncthreads();

  // q = own_e @ Wq : K=128 (A fragments hoisted across the 2 H-tiles)
  {
    v16h afr[4];
#pragma unroll
    for (int kc = 0; kc < 4; ++kc) afr[kc] = frag_ld(oeh, 68, 0, kc * 16, lane);
#pragma unroll
    for (int t = 0; t < 2; ++t) {
      int ht = wid + 4 * t;
      v8f acc = vzero();
#pragma unroll
      for (int kc = 0; kc < 4; ++kc)
        acc = wmma16(afr[kc], frag_ld(wql, 68, ht * 16, kc * 16, lane), acc);
      int h = ht * 16 + col;
#pragma unroll
      for (int r = 0; r < 8; ++r) {
        int row = r + 8 * half;
        q_out[(size_t)(r0 + row) * H + h] = acc[r];
      }
    }
  }

  // own_grid: K=512 in 4 chunks of 128, weight chunk staged in LDS
  v8f ga0 = vzero(), ga1 = vzero();
  for (int kc4 = 0; kc4 < 4; ++kc4) {
    __syncthreads();
    for (int i = tid; i < 128 * 64; i += 128) {
      int h = i >> 6, p = i & 63;
      wgl[h * 68 + p] = wg[h * 256 + kc4 * 64 + p];
    }
    __syncthreads();
#pragma unroll
    for (int kk = 0; kk < 4; ++kk) {
      v16h a = frag_ld(s1h, 260, 0, kc4 * 64 + kk * 16, lane);
      ga0 = wmma16(a, frag_ld(wgl, 68, (wid) * 16, kk * 16, lane), ga0);
      ga1 = wmma16(a, frag_ld(wgl, 68, (wid + 4) * 16, kk * 16, lane), ga1);
    }
  }
#pragma unroll
  for (int t = 0; t < 2; ++t) {
    int ht = wid + 4 * t;
    v8f* ga = t ? &ga1 : &ga0;
    int h = ht * 16 + col;
    float bias = b_grid[h];
#pragma unroll
    for (int r = 0; r < 8; ++r) {
      int row = r + 8 * half;
      float v = (*ga)[r] + bias; v = v > 0.f ? v : 0.f;
      grid_out[(size_t)(r0 + row) * H + h] = v;
    }
  }
}

// ---------------------------------------------------------------- attention
// one block (8 waves) per batch row b; everything fused in LDS
__global__ __launch_bounds__(256) void attn_kernel(
    const float* __restrict__ s2, const float* __restrict__ b_intr,
    const uint32_t* __restrict__ wintr, const uint32_t* __restrict__ wkT,
    const uint32_t* __restrict__ wvT, const float* __restrict__ own_e,
    const float* __restrict__ q_g, const float* __restrict__ grid_g,
    const float* __restrict__ Wc, const float* __restrict__ bcv,
    float* __restrict__ outp, int B) {
  __shared__ alignas(16) uint32_t s2h[128 * 20];   // state2 row-block f16, K=32 pad
  __shared__ alignas(16) uint32_t wil[128 * 20];
  __shared__ alignas(16) uint32_t xeh[128 * 68];   // x_e f16 [n][128]
  __shared__ alignas(16) uint32_t wkl[128 * 68];
  __shared__ alignas(16) uint32_t wvl[128 * 68];
  __shared__ float kv[128 * 133];                  // k then v, f32
  __shared__ float s2f[128 * 20];
  __shared__ float q_l[128], oe_l[128], og_l[128];
  __shared__ float msum[128], score[128], alv[128], red[64], part[128 * 4];
  __shared__ float smax, ssum;

  const int tid = threadIdx.x, lane = tid & 31, wid = tid >> 5;
  const int b = blockIdx.x;
  const int half = lane >> 4, col = lane & 15, m0 = wid * 16;

  for (int i = tid; i < 128 * 20; i += 256) s2f[i] = s2[(size_t)b * (128 * DI) + i];
  for (int i = tid; i < 128 * 16; i += 256) { int h = i >> 4, p = i & 15; wil[h * 20 + p] = wintr[i]; }
  for (int i = tid; i < 128 * 64; i += 256) {
    int h = i >> 6, p = i & 63;
    wkl[h * 68 + p] = wkT[i];
    wvl[h * 68 + p] = wvT[i];
  }
  if (tid < 128) {
    q_l[tid]  = q_g[(size_t)b * H + tid];
    oe_l[tid] = own_e[(size_t)b * H + tid];
    og_l[tid] = grid_g[(size_t)b * H + tid];
  }
  __syncthreads();

  for (int i = tid; i < 128 * 16; i += 256) {      // f32 -> f16 (pad K to 32)
    int n = i >> 4, p = i & 15, k0 = 2 * p;
    float f0 = (k0 < DI) ? s2f[n * 20 + k0] : 0.f;
    float f1 = (k0 + 1 < DI) ? s2f[n * 20 + k0 + 1] : 0.f;
    s2h[n * 20 + p] = packh(f0, f1);
  }
  if (tid < 128) {                                 // mask sums
    float s = 0.f;
    for (int k = 0; k < DI; ++k) s += s2f[tid * 20 + k];
    msum[tid] = s;
  }
  __syncthreads();

  // GEMM1: x_e = relu(s2 @ W_intr + b), f16 into LDS (A frag hoisted)
  uint16_t* xe16 = reinterpret_cast<uint16_t*>(xeh);
  {
    v16h a = frag_ld(s2h, 20, m0, 0, lane);
#pragma unroll
    for (int ht = 0; ht < 8; ++ht) {
      v8f acc = vzero();
      acc = wmma16(a, frag_ld(wil, 20, ht * 16, 0, lane), acc);
      int h = ht * 16 + col;
      float bias = b_intr[h];
#pragma unroll
      for (int r = 0; r < 8; ++r) {
        int n = m0 + r + 8 * half;
        float v = acc[r] + bias; v = v > 0.f ? v : 0.f;
        xe16[n * 136 + h] = f2h16(v);
      }
    }
  }
  __syncthreads();

  // A fragments of x_e for this wave's m0: shared by GEMM2 and GEMM3
  v16h afr[4];
#pragma unroll
  for (int kc = 0; kc < 4; ++kc) afr[kc] = frag_ld(xeh, 68, m0, kc * 16, lane);

  // GEMM2: k = x_e @ Wk -> kv (f32)
#pragma unroll
  for (int ht = 0; ht < 8; ++ht) {
    v8f acc = vzero();
#pragma unroll
    for (int kc = 0; kc < 4; ++kc)
      acc = wmma16(afr[kc], frag_ld(wkl, 68, ht * 16, kc * 16, lane), acc);
#pragma unroll
    for (int r = 0; r < 8; ++r) { int n = m0 + r + 8 * half; kv[n * 133 + ht * 16 + col] = acc[r]; }
  }
  __syncthreads();

  // score + masked softmax (deterministic fixed-order reductions)
  if (tid < 128) {
    float s = 0.f;
    for (int h = 0; h < H; ++h) s += kv[tid * 133 + h] * q_l[h];
    s *= 0.08838834764831845f;                      // 1/sqrt(128)
    score[tid] = (msum[tid] != 0.f) ? s : -3.0e38f;
  }
  __syncthreads();
  if (tid < 64) red[tid] = fmaxf(score[tid], score[tid + 64]);
  __syncthreads();
  if (tid == 0) { float m = red[0]; for (int i = 1; i < 64; ++i) m = fmaxf(m, red[i]); smax = m; }
  __syncthreads();
  if (tid < 128) alv[tid] = (msum[tid] != 0.f) ? expf(score[tid] - smax) : 0.f;
  __syncthreads();
  if (tid < 64) red[tid] = alv[tid] + alv[tid + 64];
  __syncthreads();
  if (tid == 0) { float s = 0.f; for (int i = 0; i < 64; ++i) s += red[i]; ssum = (s > 0.f) ? s : 1.f; }
  __syncthreads();

  // GEMM3: v = x_e @ Wv -> kv (overwrite; k fully consumed above)
#pragma unroll
  for (int ht = 0; ht < 8; ++ht) {
    v8f acc = vzero();
#pragma unroll
    for (int kc = 0; kc < 4; ++kc)
      acc = wmma16(afr[kc], frag_ld(wvl, 68, ht * 16, kc * 16, lane), acc);
#pragma unroll
    for (int r = 0; r < 8; ++r) { int n = m0 + r + 8 * half; kv[n * 133 + ht * 16 + col] = acc[r]; }
  }
  __syncthreads();

  // v_att + folded head: out = [own_e|v_att|own_grid] @ (W1@W2) + (b1@W2+b2)
  if (tid < 128) {
    float inv = 1.f / ssum;
    float va = 0.f;
    for (int n = 0; n < 128; ++n) va += alv[n] * kv[n * 133 + tid];
    va *= inv;
    float po = oe_l[tid], pg = og_l[tid];
#pragma unroll
    for (int j = 0; j < 4; ++j)
      part[tid * 4 + j] = po * Wc[tid * 4 + j] + va * Wc[(128 + tid) * 4 + j] +
                          pg * Wc[(256 + tid) * 4 + j];
  }
  __syncthreads();
  if (tid < 4) {
    float s = bcv[tid];
    for (int h = 0; h < 128; ++h) s += part[h * 4 + tid];
    outp[(size_t)b * 4 + tid] = s;                                       // mean
    outp[(size_t)B * 4 + (size_t)b * 4 + tid] = fminf(fmaxf(s, -20.f), 2.f); // log_std
  }
}

// ---------------------------------------------------------------- launch
extern "C" void kernel_launch(void* const* d_in, const int* in_sizes, int n_in,
                              void* d_out, int out_size, void* d_ws, size_t ws_size,
                              hipStream_t stream) {
  const float* s0     = (const float*)d_in[0];
  const float* s1     = (const float*)d_in[1];
  const float* s2     = (const float*)d_in[2];
  const float* W_own  = (const float*)d_in[3];
  const float* b_own  = (const float*)d_in[4];
  const float* W_intr = (const float*)d_in[5];
  const float* b_intr = (const float*)d_in[6];
  const float* W_grid = (const float*)d_in[7];
  const float* b_grid = (const float*)d_in[8];
  const float* Wq     = (const float*)d_in[9];
  const float* Wk     = (const float*)d_in[10];
  const float* Wv     = (const float*)d_in[11];
  const float* W1     = (const float*)d_in[12];
  const float* b1     = (const float*)d_in[13];
  const float* W2     = (const float*)d_in[14];
  const float* b2     = (const float*)d_in[15];

  const int B = in_sizes[0] / DOWN;
  uint32_t* wsw = (uint32_t*)d_ws;
  float*    wsf = (float*)d_ws;
  float* own_e    = wsf + OW_DYN;
  float* q_buf    = own_e + (size_t)B * H;
  float* grid_buf = q_buf + (size_t)B * H;

  prep_kernel<<<1, 256, 0, stream>>>(W_own, W_intr, Wq, Wk, Wv, W_grid, W1, b1, W2, b2, wsw);
  embed_kernel<<<B / 16, 128, 0, stream>>>(s0, s1, b_own, b_grid,
                                           wsw + OW_WOWN, wsw + OW_WQ, wsw + OW_WG,
                                           own_e, q_buf, grid_buf);
  attn_kernel<<<B, 256, 0, stream>>>(s2, b_intr, wsw + OW_WINTR, wsw + OW_WK, wsw + OW_WV,
                                     own_e, q_buf, grid_buf,
                                     wsf + OW_WC, wsf + OW_BC, (float*)d_out, B);
}
